// MultiHeadAttention_38517266710599
// MI455X (gfx1250) — matrix operand order repaired and checked by reading, not compiled, not run on hardware
//
#include <hip/hip_runtime.h>
#include <hip/hip_bf16.h>

typedef __bf16 bf16_t;
typedef __attribute__((ext_vector_type(4)))  float   v4f;
typedef __attribute__((ext_vector_type(8)))  float   v8f;
typedef __attribute__((ext_vector_type(4)))  __bf16  v4bf;
typedef __attribute__((ext_vector_type(8)))  __bf16  v8bf;
typedef __attribute__((ext_vector_type(16))) __bf16  v16bf;

// ---- problem constants ----
#define Bq   2
#define Sq   2048
#define Dq   1024
#define Hq   16
#define HDq  64

// ---------------------------------------------------------------------
// CDNA5 async global->LDS copy (ASYNCcnt path), with sync fallback.
// Builtin signature on this toolchain: (v4i addrspace(1)*, v4i addrspace(3)*,
// imm offset, imm cpol) — v4i spelled as gcc-style vector_size(16) int.
// ---------------------------------------------------------------------
#if defined(__gfx1250__) && __has_builtin(__builtin_amdgcn_global_load_async_to_lds_b128)
#define USE_ASYNC 1
#else
#define USE_ASYNC 0
#endif

typedef int v4i_t __attribute__((__vector_size__(16)));
typedef __attribute__((address_space(1))) v4i_t gv4i_t;
typedef __attribute__((address_space(3))) v4i_t lv4i_t;

static __device__ __forceinline__ void cp16_async(const void* g, void* l) {
#if USE_ASYNC
  // Flat LDS addresses map to the LDS offset via addr[31:0] (ISA 10.2
  // aperture rules), so the integer truncation yields the AS3 pointer.
  __builtin_amdgcn_global_load_async_to_lds_b128(
      (gv4i_t*)(uintptr_t)g, (lv4i_t*)(uint32_t)(uintptr_t)l, 0, 0);
#else
  *(uint4*)l = *(const uint4*)g;
#endif
}

static __device__ __forceinline__ void async_wait_all() {
#if USE_ASYNC
#if __has_builtin(__builtin_amdgcn_s_wait_asynccnt)
  __builtin_amdgcn_s_wait_asynccnt(0);
#else
  asm volatile("s_wait_asynccnt 0" ::: "memory");
#endif
#endif
}

static __device__ __forceinline__ v8f wmma_bf16(v16bf a, v16bf b, v8f c) {
  return __builtin_amdgcn_wmma_f32_16x16x32_bf16(false, a, false, b, (short)0, c,
                                                 false, false);
}

// A-fragment (16x32 bf16) per ISA 7.12.2 from row-major storage:
// lane holds K = kh*8 + {0..7} in elems 0..7 and K = kh*8+16 + {0..7} in 8..15.
static __device__ __forceinline__ v16bf afrag(const bf16_t* lo) {
  v8bf a = *(const v8bf*)lo;
  v8bf b = *(const v8bf*)(lo + 16);
  return __builtin_shufflevector(a, b, 0, 1, 2, 3, 4, 5, 6, 7,
                                 8, 9, 10, 11, 12, 13, 14, 15);
}

// =====================================================================
// Prepass 1: fp32 -> bf16 straight convert (x).
// =====================================================================
__global__ __launch_bounds__(256) void cvt_kernel(const float* __restrict__ src,
                                                  bf16_t* __restrict__ dst,
                                                  int n) {
  int i = (blockIdx.x * 256 + threadIdx.x) * 4;
  if (i + 3 < n) {
    v4f a = *(const v4f*)(src + i);
    v4bf o = {(bf16_t)a.x, (bf16_t)a.y, (bf16_t)a.z, (bf16_t)a.w};
    *(v4bf*)(dst + i) = o;
  }
}

// =====================================================================
// Prepass 2: transpose-convert W[k][n] (fp32) -> Wt[n][k] (bf16).
// z selects among the 4 weight matrices. 32x32 LDS tile transpose.
// =====================================================================
__global__ __launch_bounds__(256) void wtr_kernel(
    const float* __restrict__ W0, const float* __restrict__ W1,
    const float* __restrict__ W2, const float* __restrict__ W3,
    bf16_t* __restrict__ T0, bf16_t* __restrict__ T1,
    bf16_t* __restrict__ T2, bf16_t* __restrict__ T3) {
  __shared__ float t[32][33];
  const float* W = W0;
  bf16_t* T = T0;
  if (blockIdx.z == 1) { W = W1; T = T1; }
  if (blockIdx.z == 2) { W = W2; T = T2; }
  if (blockIdx.z == 3) { W = W3; T = T3; }
  const int bx = blockIdx.x * 32;  // n-range
  const int by = blockIdx.y * 32;  // k-range
  const int lx = threadIdx.x & 31;
  const int ly = threadIdx.x >> 5;
#pragma unroll
  for (int j = 0; j < 32; j += 8)
    t[ly + j][lx] = W[(size_t)(by + ly + j) * Dq + bx + lx];
  __syncthreads();
#pragma unroll
  for (int j = 0; j < 32; j += 8)
    T[(size_t)(bx + ly + j) * Dq + by + lx] = (bf16_t)t[lx][ly + j];
}

// =====================================================================
// Kernel: 128x128-tile bf16 WMMA GEMM, async LDS staging.
//  mode 0: C = xb * Wt(z)   -> z0/z1: bf16 out in [b,h,s,hd]
//                              z2   : bf16 out in [b,h,hd,s] (pre-transposed V)
//  mode 1: C = Ab * Wot + bo -> f32 out row-major
// A and B are bf16; B is pre-transposed (Bt[n][k]).
// =====================================================================
__global__ __launch_bounds__(256) void gemm128_kernel(
    const bf16_t* __restrict__ A, const bf16_t* __restrict__ Bt0,
    const bf16_t* __restrict__ Bt1, const bf16_t* __restrict__ Bt2,
    const float* __restrict__ bias, bf16_t* __restrict__ O0,
    bf16_t* __restrict__ O1, bf16_t* __restrict__ O2,
    float* __restrict__ Of, int mode) {
  const int N = 1024, K = 1024;
  alignas(32) __shared__ bf16_t As[128 * 72];  // [row][k], ldk=72 (16B-aligned rows)
  alignas(32) __shared__ bf16_t Bs[128 * 80];  // [col][k], ldk=80 (32B-aligned cols)

  const int tid  = threadIdx.x;
  const int wave = tid >> 5;
  const int lane = tid & 31;
  const int wr   = wave & 3;   // 4 row-groups of 32 rows
  const int wc   = wave >> 2;  // 2 col-groups of 64 cols
  const int bm   = blockIdx.x * 128;
  const int bn   = blockIdx.y * 128;

  const bf16_t* Bt = Bt0;
  bf16_t* Osel = O0;
  bool vtr = false;
  if (mode == 0) {
    if (blockIdx.z == 1) { Bt = Bt1; Osel = O1; }
    if (blockIdx.z == 2) { Bt = Bt2; Osel = O2; vtr = true; }
  }

  v8f acc[2][4];
#pragma unroll
  for (int i = 0; i < 2; ++i)
#pragma unroll
    for (int j = 0; j < 4; ++j) acc[i][j] = (v8f)0.0f;

  const int r  = lane & 15;
  const int kh = lane >> 4;

  for (int kb = 0; kb < K; kb += 64) {
    __syncthreads();
    // ---- stage A tile (128x64) and B tile (128x64), 16B granules ----
#pragma unroll
    for (int i = 0; i < 4; ++i) {
      int g  = tid + i * 256;     // 0..1023
      int rr = g >> 3;            // 0..127
      int c  = (g & 7) << 3;      // 0..56
      cp16_async(A + (size_t)(bm + rr) * K + kb + c, As + rr * 72 + c);
    }
#pragma unroll
    for (int i = 0; i < 4; ++i) {
      int g  = tid + i * 256;
      int nn = g >> 3;
      int c  = (g & 7) << 3;
      cp16_async(Bt + (size_t)(bn + nn) * K + kb + c, Bs + nn * 80 + c);
    }
    async_wait_all();
    __syncthreads();

#pragma unroll
    for (int ks = 0; ks < 64; ks += 32) {
      v16bf af[2], bfr[4];
#pragma unroll
      for (int ti = 0; ti < 2; ++ti)
        af[ti] = afrag(As + (wr * 32 + ti * 16 + r) * 72 + ks + kh * 8);
#pragma unroll
      for (int tj = 0; tj < 4; ++tj)
        bfr[tj] = afrag(Bs + (wc * 64 + tj * 16 + r) * 80 + ks + kh * 8);
#pragma unroll
      for (int ti = 0; ti < 2; ++ti)
#pragma unroll
        for (int tj = 0; tj < 4; ++tj)
          acc[ti][tj] = wmma_bf16(af[ti], bfr[tj], acc[ti][tj]);
    }
  }

  // ---- epilogue: C fragment element = (M = i + kh*8, N = lane&15) ----
#pragma unroll
  for (int ti = 0; ti < 2; ++ti)
#pragma unroll
    for (int tj = 0; tj < 4; ++tj)
#pragma unroll
      for (int i = 0; i < 8; ++i) {
        int row = bm + wr * 32 + ti * 16 + kh * 8 + i;
        int col = bn + wc * 64 + tj * 16 + r;
        float v = acc[ti][tj][i];
        if (mode == 0) {
          int b = row >> 11, s = row & (Sq - 1);
          int h = col >> 6,  hd = col & (HDq - 1);
          if (vtr)  // V stored [b,h,hd,s] so attention's V-tile copy is contiguous
            Osel[((size_t)(b * Hq + h) * HDq + hd) * Sq + s] = (bf16_t)v;
          else
            Osel[(((size_t)(b * Hq + h) << 11) + s) * HDq + hd] = (bf16_t)v;
        } else {
          Of[(size_t)row * N + col] = v + bias[col];
        }
      }
}

// =====================================================================
// Causal flash attention. One (b*h) per blockIdx.y, 128 queries per
// block (16 per wave), 128-key tiles async-staged into LDS.
// V arrives pre-transposed in global ([b,h,hd,s]) -> contiguous rows.
// =====================================================================
__global__ __launch_bounds__(256) void attn_kernel(
    const bf16_t* __restrict__ Q, const bf16_t* __restrict__ Kv,
    const bf16_t* __restrict__ Vt_g, bf16_t* __restrict__ Ob) {
  alignas(32) __shared__ bf16_t Kt[128 * 64];      // [key][hd]
  alignas(32) __shared__ bf16_t Vt[64 * 128];      // [hd][key]
  alignas(32) __shared__ bf16_t Ps[8 * 16 * 128];  // per-wave P tile 16x128

  const int tid  = threadIdx.x;
  const int wave = tid >> 5;
  const int lane = tid & 31;
  const int r    = lane & 15;
  const int kh   = lane >> 4;
  const int bx   = blockIdx.x;  // query block (16)
  const int bh   = blockIdx.y;  // b*H + h     (32)
  const size_t base  = (size_t)bh * Sq * HDq;
  const size_t vbase = (size_t)bh * HDq * Sq;  // [b,h,hd,s]
  const int qBase = bx * 128 + wave * 16;

  v16bf qf[2];
  {
    const bf16_t* qp = Q + base + (size_t)(qBase + r) * HDq;
#pragma unroll
    for (int s = 0; s < 2; ++s) qf[s] = afrag(qp + s * 32 + kh * 8);
  }

  float mrow[8], lrow[8];
  v8f o[4];
#pragma unroll
  for (int i = 0; i < 8; ++i) { mrow[i] = -3.0e38f; lrow[i] = 0.0f; }
#pragma unroll
  for (int j = 0; j < 4; ++j) o[j] = (v8f)0.0f;

  for (int kb = 0; kb <= bx; ++kb) {
    __syncthreads();
    // ---- K tile: 16KB contiguous ----
#pragma unroll
    for (int i = 0; i < 4; ++i) {
      int g = tid + i * 256;  // 0..1023 granules of 8 bf16
      cp16_async(Kv + base + (size_t)kb * 128 * HDq + g * 8, Kt + g * 8);
    }
    // ---- V tile: 64 rows of 128 contiguous keys ----
#pragma unroll
    for (int i = 0; i < 4; ++i) {
      int g  = tid + i * 256;  // 0..1023
      int hd = g >> 4;
      int c  = (g & 15) << 3;  // 0..120
      cp16_async(Vt_g + vbase + (size_t)hd * Sq + kb * 128 + c,
                 Vt + hd * 128 + c);
    }
    async_wait_all();
    __syncthreads();

    // ---- scores: S(16x128) = Q(16x64) * K^T ----
    v8f sc[8];
#pragma unroll
    for (int t = 0; t < 8; ++t) sc[t] = (v8f)0.0f;
#pragma unroll
    for (int s = 0; s < 2; ++s)
#pragma unroll
      for (int t = 0; t < 8; ++t) {
        v16bf kf = afrag(Kt + (t * 16 + r) * HDq + s * 32 + kh * 8);
        sc[t] = wmma_bf16(qf[s], kf, sc[t]);
      }

    // ---- scale + causal mask (only diagonal key block overlaps) ----
    const float scale = 0.125f;  // 1/sqrt(64)
    const bool diag = (kb == bx);
#pragma unroll
    for (int t = 0; t < 8; ++t)
#pragma unroll
      for (int i = 0; i < 8; ++i) {
        float v = sc[t][i] * scale;
        if (diag) {
          int key = kb * 128 + t * 16 + r;
          int q   = qBase + kh * 8 + i;
          if (key > q) v = -1.0e30f;
        }
        sc[t][i] = v;
      }

    // ---- streaming softmax: shfl reduction within the 16-lane half ----
    float alpha[8];
#pragma unroll
    for (int i = 0; i < 8; ++i) {
      float mx = sc[0][i];
#pragma unroll
      for (int t = 1; t < 8; ++t) mx = fmaxf(mx, sc[t][i]);
#pragma unroll
      for (int m = 1; m < 16; m <<= 1) mx = fmaxf(mx, __shfl_xor(mx, m, 32));
      float mnew = fmaxf(mrow[i], mx);
      alpha[i] = __expf(mrow[i] - mnew);
      mrow[i]  = mnew;
      float sum = 0.0f;
#pragma unroll
      for (int t = 0; t < 8; ++t) {
        float p = __expf(sc[t][i] - mnew);
        sc[t][i] = p;
        sum += p;
      }
#pragma unroll
      for (int m = 1; m < 16; m <<= 1) sum += __shfl_xor(sum, m, 32);
      lrow[i] = lrow[i] * alpha[i] + sum;
    }

    // ---- write P (C-layout) to per-wave LDS, rescale O ----
    bf16_t* pw = Ps + wave * (16 * 128);
#pragma unroll
    for (int t = 0; t < 8; ++t)
#pragma unroll
      for (int i = 0; i < 8; ++i)
        pw[(kh * 8 + i) * 128 + t * 16 + r] = (bf16_t)sc[t][i];
#pragma unroll
    for (int j = 0; j < 4; ++j)
#pragma unroll
      for (int i = 0; i < 8; ++i) o[j][i] *= alpha[i];

    // ---- O += P(16x128) * V(128x64) ----
#pragma unroll
    for (int s = 0; s < 4; ++s) {
      v16bf pf = afrag(pw + r * 128 + s * 32 + kh * 8);
#pragma unroll
      for (int j = 0; j < 4; ++j) {
        v16bf vf = afrag(Vt + (j * 16 + r) * 128 + s * 32 + kh * 8);
        o[j] = wmma_bf16(pf, vf, o[j]);
      }
    }
  }

  // ---- normalize, store bf16 in [b,h,s,hd] (reference's flat layout) ----
#pragma unroll
  for (int j = 0; j < 4; ++j)
#pragma unroll
    for (int i = 0; i < 8; ++i) {
      float v = o[j][i] / lrow[i];
      int q  = qBase + kh * 8 + i;
      int hd = j * 16 + r;
      Ob[base + (size_t)q * HDq + hd] = (bf16_t)v;
    }
}

// =====================================================================
extern "C" void kernel_launch(void* const* d_in, const int* in_sizes, int n_in,
                              void* d_out, int out_size, void* d_ws,
                              size_t ws_size, hipStream_t stream) {
  const float* x  = (const float*)d_in[0];
  const float* Wq = (const float*)d_in[1];
  const float* Wk = (const float*)d_in[2];
  const float* Wv = (const float*)d_in[3];
  const float* Wo = (const float*)d_in[4];
  const float* bo = (const float*)d_in[5];
  float* out = (float*)d_out;

  const size_t nX = (size_t)Bq * Sq * Dq;  // 4,194,304
  const size_t nW = (size_t)Dq * Dq;       // 1,048,576
  bf16_t* xb  = (bf16_t*)d_ws;
  bf16_t* Qb  = xb + nX;
  bf16_t* Kb  = Qb + nX;
  bf16_t* Vtg = Kb + nX;       // V in [b,h,hd,s]
  bf16_t* Ab  = Vtg + nX;
  bf16_t* Wqt = Ab + nX;       // Wt[n][k] bf16
  bf16_t* Wkt = Wqt + nW;
  bf16_t* Wvt = Wkt + nW;
  bf16_t* Wot = Wvt + nW;

  dim3 blk(256);
  // Prepass: bf16 convert x; transpose-convert the 4 weight matrices.
  cvt_kernel<<<dim3((unsigned)(nX / 1024)), blk, 0, stream>>>(x, xb, (int)nX);
  wtr_kernel<<<dim3(32, 32, 4), blk, 0, stream>>>(Wq, Wk, Wv, Wo,
                                                  Wqt, Wkt, Wvt, Wot);
  // Fused QKV projection (z selects weight/output; z==2 writes V transposed).
  gemm128_kernel<<<dim3(32, 8, 3), blk, 0, stream>>>(
      xb, Wqt, Wkt, Wvt, nullptr, Qb, Kb, Vtg, nullptr, 0);
  // Causal flash attention over all (b,h).
  attn_kernel<<<dim3(16, 32), blk, 0, stream>>>(Qb, Kb, Vtg, Ab);
  // Output projection + bias (Ab read flat as [4096,1024], matching the
  // reference's reshape-without-head-transpose).
  gemm128_kernel<<<dim3(32, 8, 1), blk, 0, stream>>>(
      Ab, Wot, nullptr, nullptr, bo, nullptr, nullptr, nullptr, out, 1);
}
